// WeLMV3Attention_90142773609337
// MI455X (gfx1250) — matrix operand order repaired and checked by reading, not compiled, not run on hardware
//
#include <hip/hip_runtime.h>
#include <math.h>
#include <stdint.h>

// ---------------- constants (match reference) ----------------
#define SLEN 2048
#define HID  2048
#define NH   16
#define NKV  4
#define HD   128          // HID / NH
#define QKVW 3072         // (NH + 2*NKV) * HD
#define KVW  512          // NKV * HD
#define GRP  4            // NH / NKV

typedef __bf16 bf16x16 __attribute__((ext_vector_type(16)));
typedef __bf16 bf16x8  __attribute__((ext_vector_type(8)));
typedef float  v8f     __attribute__((ext_vector_type(8)));

// ---------------- WMMA helper ----------------
__device__ inline v8f wmma_bf16(bf16x16 a, bf16x16 b, v8f c) {
  // (neg_a, A, neg_b, B, c_mod, C, reuse_a, reuse_b)
  return __builtin_amdgcn_wmma_f32_16x16x32_bf16(false, a, false, b, (short)0, c,
                                                 false, false);
}

// A-fragment (16x32, 16-bit): lane m = l&15, half = l>>4.
// elems e -> K = half*16 + e (one 32 B run)
// boff = per-lane BYTE offset of first 16B chunk; second chunk at +16 B.
__device__ inline bf16x16 ld_a(const __bf16* __restrict__ p, uint32_t boff) {
  const char* q = (const char*)p;
  bf16x8 lo = *(const bf16x8*)(q + boff);
  bf16x8 hi = *(const bf16x8*)(q + boff + 16);
  return __builtin_shufflevector(lo, hi, 0, 1, 2, 3, 4, 5, 6, 7,
                                 8, 9, 10, 11, 12, 13, 14, 15);
}

// B-fragment (32x16, 16-bit) from row-major B^T (N x K source):
// lane n = l&15, half = l>>4; elems e -> K = half*16 + e (one 32 B run)
__device__ inline bf16x16 ld_b(const __bf16* __restrict__ p, uint32_t boff) {
  const char* q = (const char*)p;
  bf16x8 lo = *(const bf16x8*)(q + boff);
  bf16x8 hi = *(const bf16x8*)(q + boff + 16);
  return __builtin_shufflevector(lo, hi, 0, 1, 2, 3, 4, 5, 6, 7,
                                 8, 9, 10, 11, 12, 13, 14, 15);
}

// ---------------- prep kernels ----------------
__global__ void f32_to_bf16_kernel(const float* __restrict__ x,
                                   __bf16* __restrict__ y, int n) {
  int i = blockIdx.x * blockDim.x + threadIdx.x;
  if (i < n) y[i] = (__bf16)x[i];
}

// w: K x N fp32 row-major  ->  wt: N x K bf16 row-major (transpose+convert)
__global__ void transpose_to_bf16_kernel(const float* __restrict__ w,
                                         __bf16* __restrict__ wt, int K, int N) {
  int k = blockIdx.x * blockDim.x + threadIdx.x;
  int n = blockIdx.y;
  if (k < K) wt[(size_t)n * K + k] = (__bf16)w[(size_t)k * N + n];
}

// ---------------- bf16 GEMM: C(MxN fp32) = A(MxK) * Bt(NxK)^T ----------------
// one wave per block, 64(M) x 32(N) tile per wave; K is compile-time
template <int K>
__global__ void __launch_bounds__(32, 1)
gemm_bf16_kernel(const __bf16* __restrict__ A, const __bf16* __restrict__ Bt,
                 float* __restrict__ C, int N) {
  const int nb = blockIdx.x * 32;
  const int mb = blockIdx.y * 64;
  const int lane = threadIdx.x & 31;
  const int col = lane & 15, half = lane >> 4;

  // loop-invariant 32-bit per-lane BYTE offsets (saddr + voffset form)
  uint32_t aoff[4], boff[2];
#pragma unroll
  for (int i = 0; i < 4; ++i)
    aoff[i] = ((uint32_t)(mb + 16 * i + col) * K + (uint32_t)(half * 16)) * 2u;
  boff[0] = ((uint32_t)(nb + col) * K + (uint32_t)(half * 16)) * 2u;
  boff[1] = ((uint32_t)(nb + 16 + col) * K + (uint32_t)(half * 16)) * 2u;

  v8f acc[4][2];
#pragma unroll
  for (int i = 0; i < 4; ++i)
#pragma unroll
    for (int j = 0; j < 2; ++j) acc[i][j] = (v8f){};

#pragma unroll 4
  for (int k0 = 0; k0 < K; k0 += 32) {
    const uint32_t kb = (uint32_t)k0 * 2u;     // byte advance along K
    bf16x16 b0 = ld_b(Bt, boff[0] + kb);
    bf16x16 b1 = ld_b(Bt, boff[1] + kb);
#pragma unroll
    for (int i = 0; i < 4; ++i) {
      bf16x16 ai = ld_a(A, aoff[i] + kb);
      acc[i][0] = wmma_bf16(ai, b0, acc[i][0]);
      acc[i][1] = wmma_bf16(ai, b1, acc[i][1]);
    }
  }
#pragma unroll
  for (int i = 0; i < 4; ++i) {
#pragma unroll
    for (int r = 0; r < 8; ++r) {
      int m = mb + 16 * i + half * 8 + r;
      C[(size_t)m * N + nb + col]      = acc[i][0][r];
      C[(size_t)m * N + nb + 16 + col] = acc[i][1][r];
    }
  }
}

// ---------------- RoPE on Q and K (fp32 in, bf16 out) ----------------
__global__ void rope_kernel(const float* __restrict__ qkv,
                            const int* __restrict__ pos,
                            __bf16* __restrict__ Qb, __bf16* __restrict__ Kb) {
  const int s = blockIdx.x;
  const int hh = blockIdx.y;          // 0..NH-1 = Q heads, NH..NH+NKV-1 = K heads
  const int i = threadIdx.x;          // 0..HD/2-1
  const float p = (float)pos[s];
  const float inv_freq = __expf(-((2.0f * (float)i) / (float)HD) * __logf(10000.0f));
  const float ang = p * inv_freq;
  const float c = __cosf(ang), sn = __sinf(ang);
  if (hh < NH) {
    const float* base = qkv + (size_t)s * QKVW + hh * HD;
    float x1 = base[i], x2 = base[i + HD / 2];
    __bf16* o = Qb + (size_t)s * (NH * HD) + hh * HD;
    o[i]          = (__bf16)(x1 * c - x2 * sn);
    o[i + HD / 2] = (__bf16)(x2 * c + x1 * sn);
  } else {
    const int kv = hh - NH;
    const float* base = qkv + (size_t)s * QKVW + NH * HD + kv * HD;
    float x1 = base[i], x2 = base[i + HD / 2];
    __bf16* o = Kb + (size_t)s * KVW + kv * HD;
    o[i]          = (__bf16)(x1 * c - x2 * sn);
    o[i + HD / 2] = (__bf16)(x2 * c + x1 * sn);
  }
}

// V -> bf16, stored TRANSPOSED: Vt[c][s] = V[s][c], c in [0, KVW)
__global__ void vconv_t_kernel(const float* __restrict__ qkv,
                               __bf16* __restrict__ Vt) {
  const int s = blockIdx.x * blockDim.x + threadIdx.x;
  const int c = blockIdx.y;           // kv*HD + d
  Vt[(size_t)c * SLEN + s] =
      (__bf16)qkv[(size_t)s * QKVW + NH * HD + KVW + c];
}

// ---------------- flash attention: one wave per (16-query tile, head) --------
__global__ void __launch_bounds__(32, 1)
attn_kernel(const __bf16* __restrict__ Qb, const __bf16* __restrict__ Kb,
            const __bf16* __restrict__ Vt, __bf16* __restrict__ Ctx) {
  __shared__ __bf16 ldsP[16 * 32];    // P tile staging: C-layout -> A-layout
  const int qt = blockIdx.x;          // query tile (16 rows)
  const int h  = blockIdx.y;          // head
  const int kvh = h / GRP;
  const int lane = threadIdx.x & 31;
  const int col = lane & 15, half = lane >> 4;
  const float scale = 0.08838834764831845f;  // 1/sqrt(128)

  // Q A-fragments across HD (4 x 32)
  bf16x16 aq[4];
  {
    uint32_t qoff = ((uint32_t)(qt * 16 + col) * (NH * HD) +
                     (uint32_t)(h * HD + half * 16)) * 2u;
#pragma unroll
    for (int kk = 0; kk < 4; ++kk) aq[kk] = ld_a(Qb, qoff + (uint32_t)(kk * 64));
  }

  // loop-invariant per-lane base BYTE offsets for K and V^T B-fragments
  const uint32_t kbase =
      ((uint32_t)col * KVW + (uint32_t)(kvh * HD + half * 16)) * 2u;
  const uint32_t vbase =
      ((uint32_t)(kvh * HD + col) * SLEN + (uint32_t)(half * 16)) * 2u;

  v8f O[8];
  float rmax[8], rsum[8];
#pragma unroll
  for (int j = 0; j < 8; ++j) O[j] = (v8f){};
#pragma unroll
  for (int r = 0; r < 8; ++r) { rmax[r] = -3.0e38f; rsum[r] = 0.0f; }

  const int qmax = qt * 16 + 15;
  for (int ks = 0; ks <= qmax; ks += 32) {
    const uint32_t ksK = (uint32_t)ks * (KVW * 2u);  // byte offset along keys
    // ---- scores: S(16x32) = Q(16x128) * K^T, two 16x16 WMMA column tiles
    v8f sc[2] = {};
#pragma unroll
    for (int t = 0; t < 2; ++t) {
#pragma unroll
      for (int kk = 0; kk < 4; ++kk) {
        bf16x16 bk =
            ld_b(Kb, kbase + ksK + (uint32_t)(t * 16 * KVW * 2 + kk * 64));
        sc[t] = wmma_bf16(aq[kk], bk, sc[t]);
      }
    }
    // ---- scale + causal mask (C layout: row = half*8+r, col = lane&15)
#pragma unroll
    for (int t = 0; t < 2; ++t) {
#pragma unroll
      for (int r = 0; r < 8; ++r) {
        int key = ks + t * 16 + col;
        int q   = qt * 16 + half * 8 + r;
        float sval = sc[t][r] * scale;
        sc[t][r] = (key <= q) ? sval : -3.0e38f;
      }
    }
    // ---- online softmax row statistics (reduce across the 16-lane half)
    float alpha[8];
#pragma unroll
    for (int r = 0; r < 8; ++r) {
      float m = fmaxf(sc[0][r], sc[1][r]);
      for (int off = 8; off; off >>= 1) m = fmaxf(m, __shfl_xor(m, off, 32));
      float mnew = fmaxf(rmax[r], m);
      alpha[r] = __expf(rmax[r] - mnew);
      rmax[r] = mnew;
      float p0 = __expf(sc[0][r] - mnew);
      float p1 = __expf(sc[1][r] - mnew);
      sc[0][r] = p0; sc[1][r] = p1;
      float ps = p0 + p1;
      for (int off = 8; off; off >>= 1) ps += __shfl_xor(ps, off, 32);
      rsum[r] = rsum[r] * alpha[r] + ps;
    }
    // ---- stage P (16x32) to LDS in row-major for A-fragment reload
#pragma unroll
    for (int t = 0; t < 2; ++t)
#pragma unroll
      for (int r = 0; r < 8; ++r)
        ldsP[(half * 8 + r) * 32 + t * 16 + col] = (__bf16)sc[t][r];
    // ---- rescale accumulators
#pragma unroll
    for (int j = 0; j < 8; ++j)
#pragma unroll
      for (int r = 0; r < 8; ++r)
        O[j][r] *= alpha[r];
    // ---- reload P as an A-fragment via two 16B DS vector loads per lane
    bf16x16 ap;
    {
      const int row = col;  // lane & 15
      const __bf16* lp = &ldsP[row * 32 + half * 16];
      bf16x8 lo = *(const bf16x8*)(lp);
      bf16x8 hi = *(const bf16x8*)(lp + 8);
      ap = __builtin_shufflevector(lo, hi, 0, 1, 2, 3, 4, 5, 6, 7,
                                   8, 9, 10, 11, 12, 13, 14, 15);
    }
    // ---- context: O(16x128) += P(16x32) * V(32x128); V^T rows contiguous
#pragma unroll
    for (int j = 0; j < 8; ++j) {
      bf16x16 bv = ld_b(Vt, vbase + (uint32_t)(j * 16 * SLEN * 2) +
                                (uint32_t)(ks * 2));
      O[j] = wmma_bf16(ap, bv, O[j]);
    }
  }
  // ---- epilogue: normalize and store bf16 ctx
#pragma unroll
  for (int r = 0; r < 8; ++r) {
    float inv = 1.0f / rsum[r];
    int q = qt * 16 + half * 8 + r;
#pragma unroll
    for (int j = 0; j < 8; ++j)
      Ctx[(size_t)q * (NH * HD) + h * HD + j * 16 + col] = (__bf16)(O[j][r] * inv);
  }
}

// ---------------- launch ----------------
extern "C" void kernel_launch(void* const* d_in, const int* in_sizes, int n_in,
                              void* d_out, int out_size, void* d_ws, size_t ws_size,
                              hipStream_t stream) {
  const float* hs   = (const float*)d_in[0];
  const float* Wqkv = (const float*)d_in[1];
  const float* Wo   = (const float*)d_in[2];
  const int*   pos  = (const int*)d_in[3];
  float* out = (float*)d_out;

  char* ws = (char*)d_ws;
  size_t off = 0;
  auto carve = [&](size_t bytes) {
    void* p = ws + off;
    off += (bytes + 255) & ~(size_t)255;
    return p;
  };
  __bf16* hsb    = (__bf16*)carve((size_t)SLEN * HID * 2);
  __bf16* WqkvT  = (__bf16*)carve((size_t)QKVW * HID * 2);
  __bf16* WoT    = (__bf16*)carve((size_t)HID * HID * 2);
  float*  qkvf   = (float*) carve((size_t)SLEN * QKVW * 4);
  __bf16* Qb     = (__bf16*)carve((size_t)SLEN * NH * HD * 2);
  __bf16* Kb     = (__bf16*)carve((size_t)SLEN * KVW * 2);
  __bf16* Vt     = (__bf16*)carve((size_t)KVW * SLEN * 2);
  __bf16* Ctxb   = (__bf16*)carve((size_t)SLEN * NH * HD * 2);

  // 1) operand conversion
  f32_to_bf16_kernel<<<(SLEN * HID + 255) / 256, 256, 0, stream>>>(hs, hsb, SLEN * HID);
  transpose_to_bf16_kernel<<<dim3(HID / 256, QKVW), 256, 0, stream>>>(Wqkv, WqkvT, HID, QKVW);
  transpose_to_bf16_kernel<<<dim3(HID / 256, HID), 256, 0, stream>>>(Wo, WoT, HID, HID);

  // 2) QKV projection: qkvf = hs @ Wqkv
  gemm_bf16_kernel<HID><<<dim3(QKVW / 32, SLEN / 64), 32, 0, stream>>>(hsb, WqkvT,
                                                                       qkvf, QKVW);
  // 3) RoPE + transposed V conversion
  rope_kernel<<<dim3(SLEN, NH + NKV), HD / 2, 0, stream>>>(qkvf, pos, Qb, Kb);
  vconv_t_kernel<<<dim3(SLEN / 256, KVW), 256, 0, stream>>>(qkvf, Vt);

  // 4) causal GQA flash attention
  attn_kernel<<<dim3(SLEN / 16, NH), 32, 0, stream>>>(Qb, Kb, Vt, Ctxb);

  // 5) output projection: out = ctx @ Wo
  gemm_bf16_kernel<HID><<<dim3(HID / 32, SLEN / 64), 32, 0, stream>>>(Ctxb, WoT,
                                                                      out, HID);
}
